// MLA_19258633356206
// MI455X (gfx1250) — matrix operand order repaired and checked by reading, not compiled, not run on hardware
//
#include <hip/hip_runtime.h>
#include <hip/hip_bf16.h>
#include <math.h>

typedef __bf16 bf16;
typedef __attribute__((ext_vector_type(16))) __bf16 bf16x16;
typedef __attribute__((ext_vector_type(8)))  __bf16 bf16x8;
typedef __attribute__((ext_vector_type(8)))  float  f32x8;
typedef int v4i __attribute__((vector_size(4 * sizeof(int))));
typedef __attribute__((address_space(1))) v4i* gv4i_p;
typedef __attribute__((address_space(3))) v4i* lv4i_p;

#define C_DIM    2048
#define C_HEADS  16
#define C_QRANK  1536
#define C_NOPE   128
#define C_ROPE   64
#define C_QKHD   192
#define C_B      2
#define C_S      2048
#define C_M      (C_B * C_S)        // 4096 tokens
#define C_MH     (C_M * C_HEADS)    // 65536
#define C_SCALE  0.07216878364870323f  // 192^-0.5
#define C_EPS    1e-6f

#if __has_builtin(__builtin_amdgcn_global_load_async_to_lds_b128)
#define ATTN_ASYNC 1
#endif

__device__ __forceinline__ void wait_async0() {
#if __has_builtin(__builtin_amdgcn_s_wait_asynccnt)
  __builtin_amdgcn_s_wait_asynccnt(0);
#else
  asm volatile("s_wait_asynccnt 0" ::: "memory");
#endif
}

__device__ __forceinline__ f32x8 wmma_bf16f32(bf16x16 a, bf16x16 b, f32x8 c) {
  return __builtin_amdgcn_wmma_f32_16x16x32_bf16(false, a, false, b, (short)0, c,
                                                 false, false);
}

// A fragment 16x32 bf16: lanes 0-15 hold row M=lane, K {0..7,16..23};
// lanes 16-31 hold same rows, K {8..15,24..31}. Two 16B contiguous loads.
__device__ __forceinline__ bf16x16 frag_a(const bf16* base, int lda, int row0,
                                          int k0, int lane) {
  int m = row0 + (lane & 15);
  int off = (lane < 16) ? 0 : 8;
  const bf16* p = base + (size_t)m * (size_t)lda + k0 + off;
  bf16x8 lo = *(const bf16x8*)(p);
  bf16x8 hi = *(const bf16x8*)(p + 16);
  bf16x16 r;
#pragma unroll
  for (int i = 0; i < 8; ++i) { r[i] = lo[i]; r[8 + i] = hi[i]; }
  return r;
}

// B fragment 32x16 bf16 (B[k][n] = W[n][k], W row-major in k): lane holds
// column n = lane&15; lanes 0-15 K {0..7,16..23}, lanes 16-31 K {8..15,24..31}.
__device__ __forceinline__ bf16x16 frag_b(const bf16* base, int ldb, int col0,
                                          int k0, int lane, int nmax) {
  int n = col0 + (lane & 15);
  if (n > nmax) n = nmax;
  int off = (lane < 16) ? 0 : 8;
  const bf16* p = base + (size_t)n * (size_t)ldb + k0 + off;
  bf16x8 lo = *(const bf16x8*)(p);
  bf16x8 hi = *(const bf16x8*)(p + 16);
  bf16x16 r;
#pragma unroll
  for (int i = 0; i < 8; ++i) { r[i] = lo[i]; r[8 + i] = hi[i]; }
  return r;
}

__global__ void k_cvt(const float* __restrict__ s, bf16* __restrict__ d, int n) {
  int i = blockIdx.x * blockDim.x + threadIdx.x;
  if (i < n) d[i] = (bf16)s[i];
}

// C = A(MxK, bf16) * W(NxK, bf16)^T + bias*bscale.  128x128 block tile,
// 8 waves, each wave 32x64 (2x4 WMMA tiles).  M, and M-tiles, multiples of 128.
__global__ __launch_bounds__(256) void k_gemm(
    const bf16* __restrict__ A, int lda,
    const bf16* __restrict__ W, int ldw,
    const float* __restrict__ bias, float bscale,
    float* __restrict__ Cf, bf16* __restrict__ Cb, int ldc,
    int M, int N, int K)
{
  const int lane = threadIdx.x & 31;
  const int wave = threadIdx.x >> 5;
  const int m0 = blockIdx.y * 128 + (wave & 3) * 32;
  const int n0 = blockIdx.x * 128 + (wave >> 2) * 64;
  f32x8 c[2][4] = {};
  for (int k0 = 0; k0 < K; k0 += 32) {
    bf16x16 a0 = frag_a(A, lda, m0,      k0, lane);
    bf16x16 a1 = frag_a(A, lda, m0 + 16, k0, lane);
#pragma unroll
    for (int j = 0; j < 4; ++j) {
      bf16x16 bj = frag_b(W, ldw, n0 + 16 * j, k0, lane, N - 1);
      c[0][j] = wmma_bf16f32(a0, bj, c[0][j]);
      c[1][j] = wmma_bf16f32(a1, bj, c[1][j]);
    }
  }
  const int rbase = (lane >= 16) ? 8 : 0;
#pragma unroll
  for (int i = 0; i < 2; ++i)
#pragma unroll
    for (int j = 0; j < 4; ++j) {
      int col = n0 + 16 * j + (lane & 15);
      if (col >= N) continue;
      float bv = bias ? bias[col] * bscale : 0.0f;
#pragma unroll
      for (int r = 0; r < 8; ++r) {
        int row = m0 + 16 * i + rbase + r;
        float v = c[i][j][r] + bv;
        if (Cb != nullptr) Cb[(size_t)row * ldc + col] = (bf16)v;
        else               Cf[(size_t)row * ldc + col] = v;
      }
    }
}

// Row-wise RMS norm (bias already folded in by GEMM epilogue), bf16 out.
__global__ __launch_bounds__(256) void k_rms(const float* __restrict__ in,
                                             const float* __restrict__ gamma,
                                             bf16* __restrict__ out, int C)
{
  const int row = blockIdx.x;
  const float* p = in + (size_t)row * C;
  float ss = 0.0f;
  for (int j = threadIdx.x; j < C; j += 256) { float v = p[j]; ss += v * v; }
#pragma unroll
  for (int o = 16; o >= 1; o >>= 1) ss += __shfl_xor(ss, o, 32);
  __shared__ float red[8];
  if ((threadIdx.x & 31) == 0) red[threadIdx.x >> 5] = ss;
  __syncthreads();
  if (threadIdx.x == 0) {
    float t = 0.0f;
    for (int i = 0; i < 8; ++i) t += red[i];
    red[0] = rsqrtf(t / (float)C + C_EPS);
  }
  __syncthreads();
  const float rs = red[0];
  for (int j = threadIdx.x; j < C; j += 256)
    out[(size_t)row * C + j] = (bf16)(p[j] * rs * gamma[j]);
}

// kv row: RMS over 192, *gamma, RoPE last 64 -> keys (bf16)
__global__ __launch_bounds__(256) void k_kvprep(
    const float* __restrict__ kva, const float* __restrict__ gamma,
    const float* __restrict__ freqs, bf16* __restrict__ keys)
{
  const int row = blockIdx.x;
  const int s = row & (C_S - 1);
  const int t = threadIdx.x;
  __shared__ float vals[C_QKHD];
  __shared__ float red[8];
  float v = 0.0f, ss = 0.0f;
  if (t < C_QKHD) { v = kva[(size_t)row * C_QKHD + t]; ss = v * v; }
#pragma unroll
  for (int o = 16; o >= 1; o >>= 1) ss += __shfl_xor(ss, o, 32);
  if ((t & 31) == 0) red[t >> 5] = ss;
  __syncthreads();
  if (t == 0) {
    float tot = 0.0f;
    for (int i = 0; i < 8; ++i) tot += red[i];
    red[0] = rsqrtf(tot / (float)C_QKHD + C_EPS);
  }
  __syncthreads();
  const float rs = red[0];
  if (t < C_QKHD) vals[t] = v * rs * gamma[t];
  __syncthreads();
  if (t < C_QKHD) {
    float ov;
    if (t < C_NOPE) ov = vals[t];
    else {
      const int j = t - C_NOPE, i2 = j >> 1;
      const float cs = freqs[((size_t)s * 32 + i2) * 2 + 0];
      const float sn = freqs[((size_t)s * 32 + i2) * 2 + 1];
      const float a  = vals[C_NOPE + 2 * i2];
      const float bb = vals[C_NOPE + 2 * i2 + 1];
      ov = (j & 1) ? (a * sn + bb * cs) : (a * cs - bb * sn);
    }
    keys[(size_t)row * C_QKHD + t] = (bf16)ov;
  }
}

// RoPE on q_pe: q (bf16, [m][h*192+128..191]) -> qpe (bf16, [m*16+h][64])
__global__ void k_qrope(const bf16* __restrict__ qb, const float* __restrict__ freqs,
                        bf16* __restrict__ qpe)
{
  int idx = blockIdx.x * blockDim.x + threadIdx.x;  // M*16*32 pairs
  if (idx >= C_M * C_HEADS * 32) return;
  const int i = idx & 31;
  const int h = (idx >> 5) & 15;
  const int m = idx >> 9;
  const int s = m & (C_S - 1);
  const bf16* q = qb + (size_t)m * 3072 + h * C_QKHD + C_NOPE;
  const float a  = (float)q[2 * i];
  const float bq = (float)q[2 * i + 1];
  const float cs = freqs[((size_t)s * 32 + i) * 2 + 0];
  const float sn = freqs[((size_t)s * 32 + i) * 2 + 1];
  bf16* o = qpe + ((size_t)m * C_HEADS + h) * C_ROPE;
  o[2 * i]     = (bf16)(a * cs - bq * sn);
  o[2 * i + 1] = (bf16)(a * sn + bq * cs);
}

// Flash attention.  Grid: x = b*(S/128)+qtile, y = head.  8 waves, each wave
// owns 16 query rows; key tiles of 32 staged in LDS (row-major for QK^T,
// transposed nope copy for P*V).  Key tile staged via async-to-LDS when the
// toolchain exposes the builtin; next tile prefetched into L2.
__global__ __launch_bounds__(256) void k_attn(
    const bf16* __restrict__ qabs,   // [M*16][128]
    const bf16* __restrict__ qpe,    // [M*16][64]
    const bf16* __restrict__ keys,   // [M][192]
    float* __restrict__ oheads)      // [M*16][128]
{
  const int lane = threadIdx.x & 31;
  const int wave = threadIdx.x >> 5;
  const int h  = blockIdx.y;
  const int b  = blockIdx.x / (C_S / 128);
  const int qt = blockIdx.x % (C_S / 128);
  const int s0w = qt * 128 + wave * 16;   // wave base position
  const int m0w = b * C_S + s0w;          // wave base token row

  __shared__ __attribute__((aligned(16))) bf16 kt_row[32][200];
  __shared__ __attribute__((aligned(16))) bf16 kt_col[128][40];
  __shared__ __attribute__((aligned(16))) bf16 pbuf[8][16 * 32];

  bf16x16 aq[6];
#pragma unroll
  for (int kk = 0; kk < 4; ++kk)
    aq[kk] = frag_a(qabs + h * C_NOPE, C_HEADS * C_NOPE, m0w, kk * 32, lane);
#pragma unroll
  for (int kk = 0; kk < 2; ++kk)
    aq[4 + kk] = frag_a(qpe + h * C_ROPE, C_HEADS * C_ROPE, m0w, kk * 32, lane);

  f32x8 o[8] = {};
  float mrow[8], lrow[8];
#pragma unroll
  for (int r = 0; r < 8; ++r) { mrow[r] = -3.0e38f; lrow[r] = 0.0f; }

  const int rbase = (lane >= 16) ? 8 : 0;
  const int pmax = qt * 128 + 127;
  for (int t0 = 0; t0 <= pmax; t0 += 32) {
    const size_t gbase = ((size_t)(b * C_S + t0)) * C_QKHD;  // element offset
    __syncthreads();
#if defined(ATTN_ASYNC)
    // Async DMA of key rows 0-3 (4x192 bf16) into LDS (96 x 16B chunks).
    if (threadIdx.x < 96) {
      const int e8 = threadIdx.x * 8;        // 8-element (16B) chunk
      const int kr = e8 / C_QKHD;
      const int kc = e8 - kr * C_QKHD;
      __builtin_amdgcn_global_load_async_to_lds_b128(
          (gv4i_p)(keys + gbase + e8),
          (lv4i_p)&kt_row[kr][kc], 0, 0);
    }
    if (t0 + 32 <= pmax && threadIdx.x < 96)   // L2 prefetch of next tile
      __builtin_prefetch(keys + gbase + 32 * C_QKHD + threadIdx.x * 64, 0, 1);
    wait_async0();
    __syncthreads();
    // Build transposed nope copy from the row-major tile (LDS -> LDS).
    for (int e = threadIdx.x; e < 32 * C_NOPE; e += 256) {
      const int kr = e >> 7, kc = e & 127;
      kt_col[kc][kr] = kt_row[kr][kc];
    }
#else
    for (int e = threadIdx.x; e < 32 * C_QKHD; e += 256) {
      const int kr = e / C_QKHD, kc = e - kr * C_QKHD;
      bf16 v = keys[gbase + (size_t)kr * C_QKHD + kc];
      kt_row[kr][kc] = v;
      if (kc < C_NOPE) kt_col[kc][kr] = v;
    }
    if (t0 + 32 <= pmax && threadIdx.x < 96)
      __builtin_prefetch(keys + gbase + 32 * C_QKHD + threadIdx.x * 64, 0, 1);
#endif
    __syncthreads();

    f32x8 sc0 = {}, sc1 = {};
#pragma unroll
    for (int kk = 0; kk < 4; ++kk) {
      sc0 = wmma_bf16f32(aq[kk], frag_b(&kt_row[0][0], 200, 0,  kk * 32, lane, 1 << 20), sc0);
      sc1 = wmma_bf16f32(aq[kk], frag_b(&kt_row[0][0], 200, 16, kk * 32, lane, 1 << 20), sc1);
    }
#pragma unroll
    for (int kk = 0; kk < 2; ++kk) {
      sc0 = wmma_bf16f32(aq[4 + kk], frag_b(&kt_row[0][0], 200, 0,  C_NOPE + kk * 32, lane, 1 << 20), sc0);
      sc1 = wmma_bf16f32(aq[4 + kk], frag_b(&kt_row[0][0], 200, 16, C_NOPE + kk * 32, lane, 1 << 20), sc1);
    }

    const int cp0 = t0 + (lane & 15);
    const int cp1 = cp0 + 16;
    float p0[8], p1[8];
#pragma unroll
    for (int r = 0; r < 8; ++r) {
      const int rp = s0w + rbase + r;
      float x0 = sc0[r] * C_SCALE; if (cp0 > rp) x0 = -1.0e9f;
      float x1 = sc1[r] * C_SCALE; if (cp1 > rp) x1 = -1.0e9f;
      float mx = fmaxf(x0, x1);
#pragma unroll
      for (int off2 = 8; off2 >= 1; off2 >>= 1) mx = fmaxf(mx, __shfl_xor(mx, off2, 32));
      const float mt = fmaxf(mrow[r], mx);
      const float alpha = __expf(mrow[r] - mt);
      mrow[r] = mt;
      const float e0 = __expf(x0 - mt);
      const float e1 = __expf(x1 - mt);
      float rs = e0 + e1;
#pragma unroll
      for (int off2 = 8; off2 >= 1; off2 >>= 1) rs += __shfl_xor(rs, off2, 32);
      lrow[r] = lrow[r] * alpha + rs;
#pragma unroll
      for (int f = 0; f < 8; ++f) o[f][r] *= alpha;
      p0[r] = e0; p1[r] = e1;
    }

    // P (C-layout) -> LDS -> A-layout fragment, then O += P * V
    bf16* pw = &pbuf[wave][0];
#pragma unroll
    for (int r = 0; r < 8; ++r) {
      pw[(rbase + r) * 32 + (lane & 15)]      = (bf16)p0[r];
      pw[(rbase + r) * 32 + (lane & 15) + 16] = (bf16)p1[r];
    }
    asm volatile("s_wait_dscnt 0" ::: "memory");
    bf16x16 pa = frag_a(pw, 32, 0, 0, lane);
#pragma unroll
    for (int f = 0; f < 8; ++f)
      o[f] = wmma_bf16f32(pa, frag_b(&kt_col[0][0], 40, f * 16, 0, lane, 1 << 20), o[f]);
  }

#pragma unroll
  for (int r = 0; r < 8; ++r) {
    const float inv = 1.0f / lrow[r];
    const size_t mr = (size_t)(m0w + rbase + r) * C_HEADS + h;
#pragma unroll
    for (int f = 0; f < 8; ++f)
      oheads[mr * C_NOPE + f * 16 + (lane & 15)] = o[f][r] * inv;
  }
}

__global__ void k_headsum(const float* __restrict__ oh, bf16* __restrict__ acc) {
  int i = blockIdx.x * blockDim.x + threadIdx.x;  // M*128
  if (i >= C_M * C_NOPE) return;
  int m = i >> 7, d = i & 127;
  float s = 0.0f;
#pragma unroll
  for (int h = 0; h < C_HEADS; ++h)
    s += oh[(((size_t)m * C_HEADS) + h) * C_NOPE + d];
  acc[i] = (bf16)s;
}

extern "C" void kernel_launch(void* const* d_in, const int* in_sizes, int n_in,
                              void* d_out, int out_size, void* d_ws, size_t ws_size,
                              hipStream_t stream)
{
  (void)in_sizes; (void)n_in; (void)out_size; (void)ws_size;
  const float* x       = (const float*)d_in[0];
  const float* freqs   = (const float*)d_in[2];
  const float* wqa_w   = (const float*)d_in[5];
  const float* wqa_bv  = (const float*)d_in[6];
  const float* wqb_w   = (const float*)d_in[7];
  const float* wqb_bv  = (const float*)d_in[8];
  const float* qnw     = (const float*)d_in[9];
  const float* wkva_w  = (const float*)d_in[10];
  const float* wkva_bv = (const float*)d_in[11];
  const float* knw     = (const float*)d_in[12];
  const float* wuk_w   = (const float*)d_in[13];
  const float* wuk_bv  = (const float*)d_in[14];
  const float* wuv_w   = (const float*)d_in[15];
  const float* wuv_bv  = (const float*)d_in[16];
  float* out = (float*)d_out;

  char* base = (char*)d_ws;
  size_t off = 0;
  auto alloc = [&](size_t bytes) -> void* {
    void* p = base + off;
    off += (bytes + 255) & ~(size_t)255;
    return p;
  };
  bf16*  xb    = (bf16*) alloc((size_t)C_M * C_DIM * 2);
  bf16*  wqaB  = (bf16*) alloc((size_t)C_QRANK * C_DIM * 2);
  bf16*  wqbB  = (bf16*) alloc((size_t)3072 * C_QRANK * 2);
  bf16*  wkvaB = (bf16*) alloc((size_t)C_QKHD * C_DIM * 2);
  bf16*  wukB  = (bf16*) alloc((size_t)C_NOPE * C_NOPE * 2);
  bf16*  wuvB  = (bf16*) alloc((size_t)C_DIM * C_NOPE * 2);
  float* qaF   = (float*)alloc((size_t)C_M * C_QRANK * 4);
  bf16*  qlatB = (bf16*) alloc((size_t)C_M * C_QRANK * 2);
  bf16*  qB    = (bf16*) alloc((size_t)C_M * 3072 * 2);
  bf16*  qpeB  = (bf16*) alloc((size_t)C_MH * C_ROPE * 2);
  bf16*  qabsB = (bf16*) alloc((size_t)C_MH * C_NOPE * 2);
  float* kvaF  = (float*)alloc((size_t)C_M * C_QKHD * 4);
  bf16*  keysB = (bf16*) alloc((size_t)C_M * C_QKHD * 2);
  float* ohF   = (float*)alloc((size_t)C_MH * C_NOPE * 4);
  bf16*  accB  = (bf16*) alloc((size_t)C_M * C_NOPE * 2);

  auto cvt = [&](const float* s, bf16* d, size_t n) {
    k_cvt<<<dim3((unsigned)((n + 255) / 256)), 256, 0, stream>>>(s, d, (int)n);
  };
  cvt(x,      xb,    (size_t)C_M * C_DIM);
  cvt(wqa_w,  wqaB,  (size_t)C_QRANK * C_DIM);
  cvt(wqb_w,  wqbB,  (size_t)3072 * C_QRANK);
  cvt(wkva_w, wkvaB, (size_t)C_QKHD * C_DIM);
  cvt(wuk_w,  wukB,  (size_t)C_NOPE * C_NOPE);
  cvt(wuv_w,  wuvB,  (size_t)C_DIM * C_NOPE);

  // q_a = x @ Wqa^T + b  (f32 out for RMS)
  k_gemm<<<dim3(C_QRANK / 128, C_M / 128), 256, 0, stream>>>(
      xb, C_DIM, wqaB, C_DIM, wqa_bv, 1.0f, qaF, nullptr, C_QRANK,
      C_M, C_QRANK, C_DIM);
  k_rms<<<C_M, 256, 0, stream>>>(qaF, qnw, qlatB, C_QRANK);
  // q = q_lat @ Wqb^T + b  (bf16 out)
  k_gemm<<<dim3(3072 / 128, C_M / 128), 256, 0, stream>>>(
      qlatB, C_QRANK, wqbB, C_QRANK, wqb_bv, 1.0f, nullptr, qB, 3072,
      C_M, 3072, C_QRANK);
  // kv_a = x @ Wkva^T + b
  k_gemm<<<dim3((C_QKHD + 127) / 128, C_M / 128), 256, 0, stream>>>(
      xb, C_DIM, wkvaB, C_DIM, wkva_bv, 1.0f, kvaF, nullptr, C_QKHD,
      C_M, C_QKHD, C_DIM);
  k_kvprep<<<C_M, 256, 0, stream>>>(kvaF, knw, freqs, keysB);
  k_qrope<<<(C_M * C_HEADS * 32 + 255) / 256, 256, 0, stream>>>(qB, freqs, qpeB);
  // q_abs = q_nope @ Wuk^T   (q_nope rows live inside qB with stride 192)
  k_gemm<<<dim3(1, C_MH / 128), 256, 0, stream>>>(
      qB, C_QKHD, wukB, C_NOPE, wuk_bv, 1.0f, nullptr, qabsB, C_NOPE,
      C_MH, C_NOPE, C_NOPE);
  // attention
  k_attn<<<dim3(C_B * (C_S / 128), C_HEADS), 256, 0, stream>>>(qabsB, qpeB, keysB, ohF);
  k_headsum<<<(C_M * C_NOPE + 255) / 256, 256, 0, stream>>>(ohF, accB);
  // out = (sum_h o_h) @ Wuv^T + 16*b
  k_gemm<<<dim3(C_DIM / 128, C_M / 128), 256, 0, stream>>>(
      accB, C_NOPE, wuvB, C_NOPE, wuv_bv, 16.0f, out, nullptr, C_DIM,
      C_M, C_DIM, C_NOPE);
}
